// ConformerMHA_51067161150011
// MI455X (gfx1250) — matrix operand order repaired and checked by reading, not compiled, not run on hardware
//
#include <hip/hip_runtime.h>
#include <hip/hip_bf16.h>
#include <stdint.h>
#include <stddef.h>

// Problem constants (match reference)
#define BB 4
#define TT 2048
#define DD 1024
#define HH 16
#define DKK 64
#define TD3 (3 * DD)
#define BT (BB * TT)

typedef __bf16 bf16;
typedef __attribute__((ext_vector_type(16))) __bf16 v16bf;
typedef __attribute__((ext_vector_type(8)))  float  v8f;
typedef __attribute__((ext_vector_type(4)))  int    v4i;

#define AS1 __attribute__((address_space(1)))
#define AS3 __attribute__((address_space(3)))

#if __has_builtin(__builtin_amdgcn_global_load_async_to_lds_b128) && \
    __has_builtin(__builtin_amdgcn_s_wait_asynccnt)
#define USE_ASYNC 1
#else
#define USE_ASYNC 0
#endif

// Copy 16 bytes (8 bf16) global -> LDS, per lane.
static __device__ __forceinline__ void cp16(bf16* lds, const bf16* g) {
#if USE_ASYNC
  __builtin_amdgcn_global_load_async_to_lds_b128(
      (AS1 v4i*)(AS1 void*)(g), (AS3 v4i*)(AS3 void*)(lds), 0, 0);
#else
  *(uint4*)lds = *(const uint4*)g;
#endif
}
static __device__ __forceinline__ void cp_wait() {
#if USE_ASYNC
  __builtin_amdgcn_s_wait_asynccnt(0);
#endif
}

// ---------------------------------------------------------------------------
// WMMA helpers (CDNA5: D = A(16x32 bf16) x B(32x16 bf16) + C(16x16 f32))
// ---------------------------------------------------------------------------
static __device__ __forceinline__ v8f wmma_bf16(v16bf a, v16bf b, v8f c) {
  return __builtin_amdgcn_wmma_f32_16x16x32_bf16(
      /*neg_a=*/false, a, /*neg_b=*/false, b,
      /*c_mod=*/(short)0, c, /*reuse_a=*/false, /*reuse_b=*/false);
}

union FragU { v16bf v; uint4 u[2]; };

// A fragment (16x32, 16-bit): lanes 0-15 hold K 0..15, lanes 16-31
// hold K 16..31 of row M=lane&15.  rowp = &tile[M][k_base].
static __device__ __forceinline__ v16bf frag_a(const bf16* rowp, int hi) {
  FragU t;
  t.u[0] = *(const uint4*)(rowp + hi * 16);
  t.u[1] = *(const uint4*)(rowp + hi * 16 + 8);
  return t.v;
}
// B fragment (32x16, 16-bit): lane n<16 holds K=0..15 of column n, lane n+16
// holds K=16..31.  With the stationary operand transposed, 16 contiguous bf16.
static __device__ __forceinline__ v16bf frag_b(const bf16* p) {
  FragU t;
  t.u[0] = *(const uint4*)(p);
  t.u[1] = *(const uint4*)(p + 8);
  return t.v;
}

static __device__ __forceinline__ v8f vzero8() {
  v8f z = {0.f, 0.f, 0.f, 0.f, 0.f, 0.f, 0.f, 0.f};
  return z;
}

// ---------------------------------------------------------------------------
// Kernel 1: LayerNorm (f32 in) -> bf16 out.  One block per row of 1024.
// ---------------------------------------------------------------------------
__global__ __launch_bounds__(256)
void ln_cast_kernel(const float* __restrict__ x, const float* __restrict__ w,
                    const float* __restrict__ b, bf16* __restrict__ out) {
  const int row = blockIdx.x;
  const int tid = threadIdx.x;
  const float4 v = ((const float4*)(x + (size_t)row * DD))[tid];

  __shared__ float stage[8];
  float s = v.x + v.y + v.z + v.w;
  #pragma unroll
  for (int o = 16; o > 0; o >>= 1) s += __shfl_xor(s, o);
  if ((tid & 31) == 0) stage[tid >> 5] = s;
  __syncthreads();
  float tot = 0.f;
  #pragma unroll
  for (int i = 0; i < 8; ++i) tot += stage[i];
  const float mu = tot * (1.0f / (float)DD);

  float4 d;
  d.x = v.x - mu; d.y = v.y - mu; d.z = v.z - mu; d.w = v.w - mu;
  float ss = d.x * d.x + d.y * d.y + d.z * d.z + d.w * d.w;
  #pragma unroll
  for (int o = 16; o > 0; o >>= 1) ss += __shfl_xor(ss, o);
  __syncthreads();
  if ((tid & 31) == 0) stage[tid >> 5] = ss;
  __syncthreads();
  float tot2 = 0.f;
  #pragma unroll
  for (int i = 0; i < 8; ++i) tot2 += stage[i];
  const float rstd = rsqrtf(tot2 * (1.0f / (float)DD) + 1e-5f);

  const float4 wv = ((const float4*)w)[tid];
  const float4 bv = ((const float4*)b)[tid];
  bf16* o4 = out + (size_t)row * DD + tid * 4;
  o4[0] = (bf16)(d.x * rstd * wv.x + bv.x);
  o4[1] = (bf16)(d.y * rstd * wv.y + bv.y);
  o4[2] = (bf16)(d.z * rstd * wv.z + bv.z);
  o4[3] = (bf16)(d.w * rstd * wv.w + bv.w);
}

// ---------------------------------------------------------------------------
// Kernel 2: cast + transpose weights:  W[K][N] f32 -> Wt[N][K] bf16
// ---------------------------------------------------------------------------
__global__ __launch_bounds__(256)
void castT_kernel(const float* __restrict__ W, bf16* __restrict__ Wt,
                  int K, int N) {
  const int n = blockIdx.x;
  for (int k = threadIdx.x; k < K; k += blockDim.x)
    Wt[(size_t)n * K + k] = (bf16)W[(size_t)k * N + n];
}

// ---------------------------------------------------------------------------
// Kernel 3: GEMM  C[M][N] = A[M][K](bf16) * Bt[N][K]^T(bf16) + bias
// Block tile 128x128, K-step 64, 8 waves (2x4), wave tile 64x32.
// ---------------------------------------------------------------------------
template <bool OUT_BF16>
__global__ __launch_bounds__(256)
void gemm_kernel(const bf16* __restrict__ A, const bf16* __restrict__ Bt,
                 const float* __restrict__ bias, void* __restrict__ Cv,
                 int M, int N, int K) {
  __shared__ bf16 smA[128][72];   // 128 rows x 64 k (+8 pad): 144B stride
  __shared__ bf16 smB[128][72];   // 128 n-rows x 64 k (+8 pad)

  const int m0 = blockIdx.y * 128;
  const int n0 = blockIdx.x * 128;
  const int tid  = threadIdx.x;
  const int lane = tid & 31;
  const int w    = tid >> 5;
  const int wm = w >> 2;          // 0..1
  const int wn = w & 3;           // 0..3
  const int l15 = lane & 15;
  const int hi  = lane >> 4;

  v8f acc[4][2];
  #pragma unroll
  for (int i = 0; i < 4; ++i)
    #pragma unroll
    for (int j = 0; j < 2; ++j) acc[i][j] = vzero8();

  const int lrow = tid >> 1;          // 0..127
  const int lcb  = (tid & 1) * 32;    // 0 or 32

  for (int kt = 0; kt < K; kt += 64) {
    __syncthreads();
    // Stage 128x64 of A and Bt (each thread: 32 bf16 from each matrix)
    {
      const bf16* ga = A  + (size_t)(m0 + lrow) * K + kt + lcb;
      const bf16* gb = Bt + (size_t)(n0 + lrow) * K + kt + lcb;
      #pragma unroll
      for (int c = 0; c < 32; c += 8) {
        cp16(&smA[lrow][lcb + c], ga + c);
        cp16(&smB[lrow][lcb + c], gb + c);
      }
#if !USE_ASYNC
      if (kt + 64 < K) {
        __builtin_prefetch(ga + 64, 0, 0);
        __builtin_prefetch(gb + 64, 0, 0);
      }
#endif
    }
    cp_wait();
    __syncthreads();

    #pragma unroll
    for (int ks = 0; ks < 2; ++ks) {
      v16bf bfr[2];
      #pragma unroll
      for (int ni = 0; ni < 2; ++ni)
        bfr[ni] = frag_b(&smB[wn * 32 + ni * 16 + l15][ks * 32 + hi * 16]);
      #pragma unroll
      for (int mi = 0; mi < 4; ++mi) {
        v16bf afr = frag_a(&smA[wm * 64 + mi * 16 + l15][ks * 32], hi);
        acc[mi][0] = wmma_bf16(afr, bfr[0], acc[mi][0]);
        acc[mi][1] = wmma_bf16(afr, bfr[1], acc[mi][1]);
      }
    }
  }

  // Epilogue: C layout per ISA: VGPR r -> (M = r + hi*8, N = l15)
  #pragma unroll
  for (int mi = 0; mi < 4; ++mi) {
    #pragma unroll
    for (int ni = 0; ni < 2; ++ni) {
      const int col = n0 + wn * 32 + ni * 16 + l15;
      const float bv = bias[col];
      #pragma unroll
      for (int r = 0; r < 8; ++r) {
        const int row = m0 + wm * 64 + mi * 16 + r + hi * 8;
        const float val = acc[mi][ni][r] + bv;
        if (OUT_BF16)
          ((bf16*)Cv)[(size_t)row * N + col] = (bf16)val;
        else
          ((float*)Cv)[(size_t)row * N + col] = val;
      }
    }
  }
}

// ---------------------------------------------------------------------------
// Kernel 4: RoPE + head transpose.  qkv[BT][3*D] bf16 ->
//   q[B,H,T,DK] (scaled by 1/sqrt(DK)), k[B,H,T,DK], vT[B,H,DK,T]   (bf16)
// ---------------------------------------------------------------------------
__global__ __launch_bounds__(64)
void rope_kernel(const bf16* __restrict__ qkv, bf16* __restrict__ qo,
                 bf16* __restrict__ ko, bf16* __restrict__ vt) {
  const int bht = blockIdx.x;
  const int t = bht % TT;
  const int h = (bht / TT) % HH;
  const int b = bht / (TT * HH);
  const int dk = threadIdx.x;  // 0..63

  const size_t base = (size_t)(b * TT + t) * TD3;
  const int c = h * DKK + dk;
  const float q = (float)qkv[base + c];
  const float k = (float)qkv[base + DD + c];
  const float v = (float)qkv[base + 2 * DD + c];

  float qp, kp;
  if (dk < 32) {
    qp = -(float)qkv[base + c + 32];
    kp = -(float)qkv[base + DD + c + 32];
  } else {
    qp = (float)qkv[base + c - 32];
    kp = (float)qkv[base + DD + c - 32];
  }
  const int j = dk & 31;
  const float freq = (float)t * __powf(10000.f, -(float)j * (1.f / 32.f));
  float sn, cs;
  __sincosf(freq, &sn, &cs);

  const size_t ob = ((size_t)(b * HH + h) * TT + t) * DKK + dk;
  qo[ob] = (bf16)((q * cs + qp * sn) * 0.125f);   // fold 1/sqrt(64)
  ko[ob] = (bf16)(k * cs + kp * sn);
  vt[((size_t)(b * HH + h) * DKK + dk) * TT + t] = (bf16)v;
}

// ---------------------------------------------------------------------------
// Kernel 5: flash attention.  grid = (T/64, B*H), block = 256 (8 waves).
// Block owns 64 queries of one head; iterates causal 64-key tiles.
// ---------------------------------------------------------------------------
__global__ __launch_bounds__(256)
void flash_kernel(const bf16* __restrict__ qg, const bf16* __restrict__ kg,
                  const bf16* __restrict__ vtg, bf16* __restrict__ og) {
  __shared__ bf16  sQ[64][72];     // [q][dk]
  __shared__ bf16  sK[64][72];     // [key][dk]
  __shared__ bf16  sV[64][72];     // [dk][key]  (V transposed)
  __shared__ float sS[64][68];     // scores [q][key]
  __shared__ bf16  sP[64][72];     // softmaxed probs, bf16
  __shared__ float s_m[64], s_l[64], s_al[64];

  const int bh = blockIdx.y;
  const int qt = blockIdx.x;
  const int q0 = qt * 64;
  const bf16* Q  = qg  + ((size_t)bh * TT + q0) * DKK;
  const bf16* Kp = kg  + (size_t)bh * TT * DKK;
  const bf16* Vt = vtg + (size_t)bh * DKK * TT;

  const int tid  = threadIdx.x;
  const int lane = tid & 31;
  const int w    = tid >> 5;
  const int l15  = lane & 15;
  const int hi   = lane >> 4;
  const int msub  = w >> 1;   // 0..3  (16-row strip of queries)
  const int npair = w & 1;    // 0..1  (32-col strip)

  const int lr = tid >> 2;            // 0..63 staging row
  const int sg = (tid & 3) * 16;      // staging 16-elem segment

  // Load the 64x64 Q tile (each thread: 16 bf16)
  {
    const bf16* gq = Q + (size_t)lr * DKK + sg;
    cp16(&sQ[lr][sg], gq);
    cp16(&sQ[lr][sg + 8], gq + 8);
  }
  if (tid < 64) { s_m[tid] = -3.0e38f; s_l[tid] = 0.f; }

  v8f o0 = vzero8(), o1 = vzero8();
  const int ntiles = qt + 1;  // causal

  for (int kt = 0; kt < ntiles; ++kt) {
    const int k0 = kt * 64;
    __syncthreads();
    // Stage K tile [key][dk] and V tile [dk][key]
    {
      const bf16* gk = Kp + (size_t)(k0 + lr) * DKK + sg;
      cp16(&sK[lr][sg], gk);
      cp16(&sK[lr][sg + 8], gk + 8);
      const bf16* gv = Vt + (size_t)lr * TT + k0 + sg;
      cp16(&sV[lr][sg], gv);
      cp16(&sV[lr][sg + 8], gv + 8);
    }
    cp_wait();
    __syncthreads();

    // S = Q * K^T  (each wave: 2 of the 16 16x16 subtiles, 2 k-steps over DK)
    #pragma unroll
    for (int nn = 0; nn < 2; ++nn) {
      const int nsub = npair * 2 + nn;
      v8f s = vzero8();
      #pragma unroll
      for (int ks = 0; ks < 2; ++ks) {
        v16bf af = frag_a(&sQ[msub * 16 + l15][ks * 32], hi);
        v16bf bf = frag_b(&sK[nsub * 16 + l15][ks * 32 + hi * 16]);
        s = wmma_bf16(af, bf, s);
      }
      #pragma unroll
      for (int r = 0; r < 8; ++r) {
        const int mrow = msub * 16 + r + hi * 8;
        const int ncol = nsub * 16 + l15;
        float val = s[r];
        if (k0 + ncol > q0 + mrow) val = -3.0e38f;  // causal mask
        sS[mrow][ncol] = val;
      }
    }
    __syncthreads();

    // Online softmax row update (one thread per query row)
    if (tid < 64) {
      const float mo = s_m[tid];
      float mt = mo;
      #pragma unroll 8
      for (int j2 = 0; j2 < 64; ++j2) mt = fmaxf(mt, sS[tid][j2]);
      const float al = __expf(mo - mt);
      float ps = 0.f;
      #pragma unroll 8
      for (int j2 = 0; j2 < 64; ++j2) {
        const float p = __expf(sS[tid][j2] - mt);
        sP[tid][j2] = (bf16)p;
        ps += p;
      }
      s_m[tid]  = mt;
      s_l[tid]  = s_l[tid] * al + ps;
      s_al[tid] = al;
    }
    __syncthreads();

    // Rescale O and accumulate O += P * V
    #pragma unroll
    for (int r = 0; r < 8; ++r) {
      const float al = s_al[msub * 16 + r + hi * 8];
      o0[r] *= al;
      o1[r] *= al;
    }
    #pragma unroll
    for (int ks = 0; ks < 2; ++ks) {
      v16bf af  = frag_a(&sP[msub * 16 + l15][ks * 32], hi);
      v16bf b0f = frag_b(&sV[(npair * 2 + 0) * 16 + l15][ks * 32 + hi * 16]);
      v16bf b1f = frag_b(&sV[(npair * 2 + 1) * 16 + l15][ks * 32 + hi * 16]);
      o0 = wmma_bf16(af, b0f, o0);
      o1 = wmma_bf16(af, b1f, o1);
    }
  }

  // Write O / l back as bf16 into [b*T+t][h*DK+dk] layout
  const int b_ = bh / HH, h_ = bh % HH;
  #pragma unroll
  for (int r = 0; r < 8; ++r) {
    const int mrow = msub * 16 + r + hi * 8;
    const float inv = 1.f / s_l[mrow];
    const size_t rowoff = (size_t)(b_ * TT + (q0 + mrow)) * DD + h_ * DKK;
    const int c0 = npair * 32 + l15;
    og[rowoff + c0]      = (bf16)(o0[r] * inv);
    og[rowoff + c0 + 16] = (bf16)(o1[r] * inv);
  }
}

// ---------------------------------------------------------------------------
// Host-side orchestration
// ---------------------------------------------------------------------------
extern "C" void kernel_launch(void* const* d_in, const int* in_sizes, int n_in,
                              void* d_out, int out_size, void* d_ws, size_t ws_size,
                              hipStream_t stream) {
  (void)in_sizes; (void)n_in; (void)out_size; (void)ws_size;
  const float* x     = (const float*)d_in[0];
  const float* ln_w  = (const float*)d_in[1];
  const float* ln_b  = (const float*)d_in[2];
  const float* w_qkv = (const float*)d_in[3];
  const float* b_qkv = (const float*)d_in[4];
  const float* w_o   = (const float*)d_in[5];
  const float* b_o   = (const float*)d_in[6];
  float* out = (float*)d_out;

  char* p = (char*)d_ws;
  auto carve = [&](size_t bytes) -> char* {
    char* r = p;
    p += (bytes + 255) & ~(size_t)255;
    return r;
  };
  bf16* h_bf    = (bf16*)carve((size_t)BT * DD * 2);     // LN output
  bf16* wqkv_t  = (bf16*)carve((size_t)TD3 * DD * 2);    // Wqkv^T
  bf16* wo_t    = (bf16*)carve((size_t)DD * DD * 2);     // Wo^T
  bf16* qkv_bf  = (bf16*)carve((size_t)BT * TD3 * 2);    // QKV projection
  bf16* q_bf    = (bf16*)carve((size_t)BB * HH * TT * DKK * 2);
  bf16* k_bf    = (bf16*)carve((size_t)BB * HH * TT * DKK * 2);
  bf16* vt_bf   = (bf16*)carve((size_t)BB * HH * DKK * TT * 2);
  bf16* attn_bf = (bf16*)carve((size_t)BT * DD * 2);

  // 1. LayerNorm -> bf16
  ln_cast_kernel<<<BT, 256, 0, stream>>>(x, ln_w, ln_b, h_bf);

  // 2. Weight cast + transpose
  castT_kernel<<<TD3, 256, 0, stream>>>(w_qkv, wqkv_t, DD, TD3);
  castT_kernel<<<DD, 256, 0, stream>>>(w_o, wo_t, DD, DD);

  // 3. QKV GEMM: [8192,1024] x [1024,3072] -> bf16 qkv
  {
    dim3 grid(TD3 / 128, BT / 128);
    gemm_kernel<true><<<grid, 256, 0, stream>>>(h_bf, wqkv_t, b_qkv,
                                                (void*)qkv_bf, BT, TD3, DD);
  }

  // 4. RoPE + head transpose (+ fold softmax scale into Q)
  rope_kernel<<<BB * HH * TT, 64, 0, stream>>>(qkv_bf, q_bf, k_bf, vt_bf);

  // 5. Flash attention
  {
    dim3 grid(TT / 64, BB * HH);
    flash_kernel<<<grid, 256, 0, stream>>>(q_bf, k_bf, vt_bf, attn_bf);
  }

  // 6. Output projection: [8192,1024] x [1024,1024] + b_o -> f32 out
  {
    dim3 grid(DD / 128, BT / 128);
    gemm_kernel<false><<<grid, 256, 0, stream>>>(attn_bf, wo_t, b_o,
                                                 (void*)out, BT, DD, DD);
  }
}
